// TransformerBlock_64682207478595
// MI455X (gfx1250) — matrix operand order repaired and checked by reading, not compiled, not run on hardware
//
#include <hip/hip_runtime.h>
#include <hip/hip_bf16.h>

// ---------------- problem constants ----------------
#define BB    4
#define SS    512
#define MAXS_ 2048
#define DD    2048
#define NH_   32
#define NKV_  8
#define HD_   64
#define HID_  5632
#define TOK   (BB*SS)          // 2048 token rows

// ---------------- bf16 helpers ----------------
typedef __bf16 bf16_t;
typedef bf16_t bf16x16 __attribute__((ext_vector_type(16)));
typedef float  floatx8 __attribute__((ext_vector_type(8)));

union AFrag { bf16x16 v; unsigned int u[8]; };
union CFrag { floatx8 v; float f[8]; };

__device__ __forceinline__ unsigned short f32_to_bf16(float f) {
    unsigned int u = __float_as_uint(f);
    u += 0x7FFFu + ((u >> 16) & 1u);          // round-to-nearest-even
    return (unsigned short)(u >> 16);
}
__device__ __forceinline__ float bf16_to_f32(unsigned short h) {
    return __uint_as_float(((unsigned int)h) << 16);
}

__device__ __forceinline__ floatx8 wmma_bf16(bf16x16 a, bf16x16 b, floatx8 c) {
    // v_wmma_f32_16x16x32_bf16 : D = A(16x32) * B(32x16) + C
    return __builtin_amdgcn_wmma_f32_16x16x32_bf16(false, a, false, b, (short)0, c, false, false);
}

// A fragment (16x32 bf16, ISA 7.12.2): lane m = L&15; half = L>>4.
// dword j maps to K-pair index (j&3) + ((j>>2)<<3) + 4*half within a 16-dword row.
__device__ __forceinline__ void ld_afrag(AFrag& f, const unsigned int* rowbase, int half) {
#pragma unroll
    for (int j = 0; j < 8; ++j)
        f.u[j] = rowbase[(j & 3) + ((j >> 2) << 3) + (half << 2)];
}
// B fragment (32x16 bf16): lane n = L&15; K pairs are dwords (j&3) + ((j>>2)<<3) + 4*half.
__device__ __forceinline__ void ld_bfrag(AFrag& f, const unsigned int* rowbase, int half) {
#pragma unroll
    for (int j = 0; j < 8; ++j)
        f.u[j] = rowbase[(j & 3) + ((j >> 2) << 3) + (half << 2)];
}

// LDS offset of a generic __shared__ pointer: low 32 bits (aperture scheme, ISA 10.2).
__device__ __forceinline__ unsigned lds_off(const void* p) {
    return (unsigned)(unsigned long long)p;
}
// Async global->LDS, 16B per lane (tracked by ASYNCcnt). INST_OFFSET applies to
// both the LDS and global addresses (ISA 15.18 pseudocode).
__device__ __forceinline__ void async_ld_b128(unsigned lds, unsigned long long gaddr) {
    asm volatile("global_load_async_to_lds_b128 %0, %1, off" :: "v"(lds), "v"(gaddr) : "memory");
}
__device__ __forceinline__ void async_ld_b128_off16(unsigned lds, unsigned long long gaddr) {
    asm volatile("global_load_async_to_lds_b128 %0, %1, off offset:16" :: "v"(lds), "v"(gaddr) : "memory");
}

// ---------------- elementwise kernels ----------------
__global__ __launch_bounds__(256)
void cvt_bf16_kernel(const float* __restrict__ in, unsigned short* __restrict__ out, long long n) {
    long long i = (long long)blockIdx.x * blockDim.x + threadIdx.x;
    long long stride = (long long)gridDim.x * blockDim.x;
    for (; i < n; i += stride) out[i] = f32_to_bf16(in[i]);
}

__global__ __launch_bounds__(256)
void rmsnorm_kernel(const float* __restrict__ x, const float* __restrict__ w,
                    unsigned short* __restrict__ out, int Dlen) {
    __shared__ float red[256];
    const int row = blockIdx.x;
    const float* xr = x + (size_t)row * Dlen;
    float ss = 0.f;
    for (int i = threadIdx.x; i < Dlen; i += 256) { float v = xr[i]; ss += v * v; }
    red[threadIdx.x] = ss;
    __syncthreads();
    for (int s2 = 128; s2 > 0; s2 >>= 1) {
        if (threadIdx.x < s2) red[threadIdx.x] += red[threadIdx.x + s2];
        __syncthreads();
    }
    const float rs = rsqrtf(red[0] / (float)Dlen + 1e-5f);
    for (int i = threadIdx.x; i < Dlen; i += 256)
        out[(size_t)row * Dlen + i] = f32_to_bf16(xr[i] * rs * w[i]);
}

// In-place RoPE on bf16 [tok, H, 64]; scale folds 1/sqrt(HD) for Q.
__global__ __launch_bounds__(256)
void rope_kernel(unsigned short* __restrict__ t, const float* __restrict__ cosb,
                 const float* __restrict__ sinb, int H, float scale, long long npairs) {
    long long i = (long long)blockIdx.x * blockDim.x + threadIdx.x;
    if (i >= npairs) return;
    int p = (int)(i & 31);                 // pair index within head (HD/2 = 32)
    long long rest = i >> 5;
    int hh = (int)(rest % H);
    long long bs = rest / H;               // b*S + s
    unsigned short* ptr = t + ((bs * H + hh) * 64) + 2 * p;
    float a = bf16_to_f32(ptr[0]);
    float b = bf16_to_f32(ptr[1]);
    float c = cosb[bs * 32 + p];
    float s = sinb[bs * 32 + p];
    ptr[0] = f32_to_bf16((a * c - b * s) * scale);
    ptr[1] = f32_to_bf16((a * s + b * c) * scale);
}

// K cache, compact GQA layout: kc[b, pos, kvh, d] (bf16)
__global__ __launch_bounds__(256)
void build_kcache_kernel(const unsigned short* __restrict__ k_rope, const float* __restrict__ cache_k,
                         const int* __restrict__ start_pos, unsigned short* __restrict__ kc) {
    unsigned int idx = blockIdx.x * 256u + threadIdx.x;   // 2^22 total
    int d   = idx & 63;
    int kvh = (idx >> 6) & 7;
    int pos = (idx >> 9) & 2047;
    int b   = idx >> 20;
    int sp  = start_pos[b];
    unsigned short v;
    if (pos >= sp && pos < sp + SS)
        v = k_rope[(((size_t)(b * SS + (pos - sp)) * NKV_) + kvh) * 64 + d];
    else
        v = f32_to_bf16(cache_k[(size_t)(b * MAXS_ + pos) * (NH_*HD_) + kvh * 256 + d]);
    kc[idx] = v;
}

// V cache, transposed: vt[b, kvh, d, pos] (bf16) so PV B-fragments are contiguous in pos.
__global__ __launch_bounds__(256)
void build_vcacheT_kernel(const unsigned short* __restrict__ v_lin, const float* __restrict__ cache_v,
                          const int* __restrict__ start_pos, unsigned short* __restrict__ vt) {
    unsigned int idx = blockIdx.x * 256u + threadIdx.x;   // 2^22 total
    int pos = idx & 2047;
    int d   = (idx >> 11) & 63;
    int kvh = (idx >> 17) & 7;
    int b   = idx >> 20;
    int sp  = start_pos[b];
    unsigned short v;
    if (pos >= sp && pos < sp + SS)
        v = v_lin[(((size_t)(b * SS + (pos - sp)) * NKV_) + kvh) * 64 + d];
    else
        v = f32_to_bf16(cache_v[(size_t)(b * MAXS_ + pos) * (NH_*HD_) + kvh * 256 + d]);
    vt[idx] = v;
}

__global__ __launch_bounds__(256)
void silu_gate_kernel(unsigned short* __restrict__ t1, const unsigned short* __restrict__ t3, long long n) {
    long long i = (long long)blockIdx.x * blockDim.x + threadIdx.x;
    long long stride = (long long)gridDim.x * blockDim.x;
    for (; i < n; i += stride) {
        float a = bf16_to_f32(t1[i]);
        float g = a / (1.f + __expf(-a));
        t1[i] = f32_to_bf16(g * bf16_to_f32(t3[i]));
    }
}

// ---------------- tiled bf16 WMMA GEMM ----------------
// C[M,N] = A[M,K](bf16) * W[N,K]^T(bf16). EPI 0: store bf16. EPI 1: store f32 + residual.
// Block tile 64(M)x128(N), 8 waves, each wave a 32x32 C tile, K-step 32.
// Global->LDS via ASYNC loads (no VGPR staging), double-buffered: tile i+1's 3
// async ops are issued, then s_wait_asynccnt 3 waits only for tile i's ops
// (async loads complete in order), which had the whole previous compute phase.
template<int EPI>
__global__ __launch_bounds__(256)
void gemm_bf16_kernel(const unsigned int* __restrict__ A,  // bf16 pairs, [M, K/2]
                      const unsigned int* __restrict__ W,  // bf16 pairs, [N, K/2]
                      unsigned short* __restrict__ Cbf,
                      float* __restrict__ Cf,
                      const float* __restrict__ resid,
                      int M, int N, int K) {
    __shared__ unsigned int Abuf[2][64 * 16];
    __shared__ unsigned int Bbuf[2][128 * 16];
    const int Kd  = K >> 1;
    const int tid = threadIdx.x;
    const int m0  = blockIdx.y * 64;
    const int n0  = blockIdx.x * 128;

    const int ar = tid >> 2, ac = (tid & 3) << 2;   // A: 64 rows x 16 dwords, 1 b128/thread
    const int br = tid >> 1, bc = (tid & 1) << 3;   // B: 128 rows x 16 dwords, 2 b128/thread
    const unsigned long long Ap = (unsigned long long)(A + (size_t)(m0 + ar) * Kd + ac);
    const unsigned long long Wp = (unsigned long long)(W + (size_t)(n0 + br) * Kd + bc);
    unsigned aoff[2], boff[2];
#pragma unroll
    for (int i = 0; i < 2; ++i) {
        aoff[i] = lds_off(&Abuf[i][ar * 16 + ac]);
        boff[i] = lds_off(&Bbuf[i][br * 16 + bc]);
    }

    // prologue: tile 0 in flight
    async_ld_b128(aoff[0], Ap);
    async_ld_b128(boff[0], Wp);
    async_ld_b128_off16(boff[0], Wp);

    const int lane = tid & 31;
    const int half = lane >> 4, ln = lane & 15;
    const int wave = tid >> 5;
    const int wm = (wave >> 2) * 32, wn = (wave & 3) * 32;

    CFrag acc[2][2];
#pragma unroll
    for (int i = 0; i < 2; ++i)
#pragma unroll
        for (int j = 0; j < 2; ++j)
#pragma unroll
            for (int t = 0; t < 8; ++t) acc[i][j].f[t] = 0.f;

    int bufi = 0;
    for (int kk = 0; kk < K; kk += 32, bufi ^= 1) {
        const bool more = (kk + 32) < K;
        if (more) {                                  // issue next tile into other buffer
            unsigned long long ga = Ap + (unsigned long long)((kk + 32) >> 1) * 4ull;
            unsigned long long gb = Wp + (unsigned long long)((kk + 32) >> 1) * 4ull;
            async_ld_b128(aoff[bufi ^ 1], ga);
            async_ld_b128(boff[bufi ^ 1], gb);
            async_ld_b128_off16(boff[bufi ^ 1], gb);
            asm volatile("s_wait_asynccnt 0x3" ::: "memory");   // current tile done
        } else {
            asm volatile("s_wait_asynccnt 0x0" ::: "memory");
        }
        __syncthreads();                              // all waves' data visible

        AFrag af[2], bf[2];
#pragma unroll
        for (int mi = 0; mi < 2; ++mi) ld_afrag(af[mi], &Abuf[bufi][(wm + mi * 16 + ln) * 16], half);
#pragma unroll
        for (int ni = 0; ni < 2; ++ni) ld_bfrag(bf[ni], &Bbuf[bufi][(wn + ni * 16 + ln) * 16], half);
#pragma unroll
        for (int mi = 0; mi < 2; ++mi)
#pragma unroll
            for (int ni = 0; ni < 2; ++ni)
                acc[mi][ni].v = wmma_bf16(af[mi].v, bf[ni].v, acc[mi][ni].v);
        __syncthreads();                              // done reading before buffer reuse
    }

#pragma unroll
    for (int mi = 0; mi < 2; ++mi)
#pragma unroll
        for (int ni = 0; ni < 2; ++ni)
#pragma unroll
            for (int r = 0; r < 8; ++r) {
                int row = m0 + wm + mi * 16 + r + 8 * half;   // C layout: row r+8*half
                int col = n0 + wn + ni * 16 + ln;             // C layout: col = lane&15
                size_t o = (size_t)row * N + col;
                float v = acc[mi][ni].f[r];
                if (EPI == 0) Cbf[o] = f32_to_bf16(v);
                else          Cf[o]  = v + resid[o];
            }
}

// ---------------- flash attention ----------------
// Block = 4 waves; each wave: one 16-query tile of one head. 32-key chunks:
// ALL 8 fragment loads (4 K + 4 V) are issued up front into distinct registers
// so they form clauses and overlap; V is independent of softmax so it prefetches
// across the whole softmax phase. 4 WMMA (Q*K^T) + online softmax + P transposed
// via LDS into an A-fragment + 4 WMMA (P*V).
__global__ __launch_bounds__(128)
void attn_kernel(const unsigned int* __restrict__ q,   // roped*0.125 bf16 [tok, 1024 dwords]
                 const unsigned int* __restrict__ kc,  // [b,pos,kvh,32 dwords]
                 const unsigned int* __restrict__ vt,  // [b,kvh,d,1024 dwords]
                 const float* __restrict__ mask,       // [b, s, MAXS]
                 unsigned short* __restrict__ out) {   // bf16 [tok, 2048]
    __shared__ float        sS[4][16 * 32];
    __shared__ unsigned int sP[4][16 * 16];
    __shared__ float        sF[4][16];

    const int tid  = threadIdx.x;
    const int wave = tid >> 5, lane = tid & 31;
    const int half = lane >> 4, ln = lane & 15;

    const int QT = SS / 64;                       // 8 q-tiles of 64 per (b,h)
    int idx = blockIdx.x;
    int b   = idx / (NH_ * QT);
    int r0  = idx % (NH_ * QT);
    int h   = r0 / QT;
    int qt  = r0 % QT;
    int qbase = qt * 64 + wave * 16;
    int kvh = h >> 2;

    // preload Q A-fragments (d 0..31 and 32..63)
    AFrag qa[2];
    {
        int tok = b * SS + qbase + ln;
        const unsigned int* qrow = q + (size_t)tok * 1024 + h * 32;
#pragma unroll
        for (int f = 0; f < 2; ++f)
#pragma unroll
            for (int j = 0; j < 8; ++j)
                qa[f].u[j] = qrow[f * 16 + (j & 3) + ((j >> 2) << 3) + (half << 2)];
    }

    CFrag o[4];
#pragma unroll
    for (int t = 0; t < 4; ++t)
#pragma unroll
        for (int r = 0; r < 8; ++r) o[t].f[r] = 0.f;
    float mi = -3.0e38f, li = 0.f;

    const float* mrow = mask + (size_t)(b * SS + qbase + ln) * MAXS_;  // lanes<16 only
    const unsigned int* vbase = vt + (size_t)(b * NKV_ + kvh) * 64 * 1024;

    for (int kk = 0; kk < MAXS_; kk += 32) {
        // ---- issue ALL loads for this chunk first (distinct registers -> clauses)
        AFrag kb[2][2];                                   // [key-tile][d-half]
#pragma unroll
        for (int kt = 0; kt < 2; ++kt) {
            int key = kk + kt * 16 + ln;
            const unsigned int* krow = kc + ((size_t)(b * MAXS_ + key) * NKV_ + kvh) * 32;
#pragma unroll
            for (int j = 0; j < 8; ++j) {
                kb[kt][0].u[j] = krow[(j & 3) + ((j >> 2) << 3) + (half << 2)];
                kb[kt][1].u[j] = krow[16 + (j & 3) + ((j >> 2) << 3) + (half << 2)];
            }
        }
        AFrag vb[4];                                      // V B-fragments, used after softmax
#pragma unroll
        for (int t = 0; t < 4; ++t) {
            const unsigned int* vrow = vbase + (size_t)(t * 16 + ln) * 1024;
#pragma unroll
            for (int j = 0; j < 8; ++j) vb[t].u[j] = vrow[(kk >> 1) + (j & 3) + ((j >> 2) << 3) + (half << 2)];
        }

        // ---- scores: two 16x16 key tiles
        CFrag sc[2];
#pragma unroll
        for (int kt = 0; kt < 2; ++kt) {
#pragma unroll
            for (int r = 0; r < 8; ++r) sc[kt].f[r] = 0.f;
            sc[kt].v = wmma_bf16(qa[0].v, kb[kt][0].v, sc[kt].v);
            sc[kt].v = wmma_bf16(qa[1].v, kb[kt][1].v, sc[kt].v);
#pragma unroll
            for (int r = 0; r < 8; ++r)
                sS[wave][(r + 8 * half) * 32 + kt * 16 + ln] = sc[kt].f[r];
        }
        __syncthreads();

        // ---- online softmax (lanes 0..15, one query row each)
        if (lane < 16) {
            float* srow = &sS[wave][lane * 32];
            const float* mr = mrow + kk;
            float sv[32];
            float cmax = -3.0e38f;
#pragma unroll
            for (int j = 0; j < 32; ++j) { sv[j] = srow[j] + mr[j]; cmax = fmaxf(cmax, sv[j]); }
            float nm = fmaxf(mi, cmax);
            float corr = __expf(mi - nm);
            li *= corr;
#pragma unroll
            for (int j = 0; j < 16; ++j) {
                float p0 = __expf(sv[2 * j]     - nm);
                float p1 = __expf(sv[2 * j + 1] - nm);
                li += p0 + p1;
                sP[wave][lane * 16 + j] =
                    (unsigned int)f32_to_bf16(p0) | ((unsigned int)f32_to_bf16(p1) << 16);
            }
            mi = nm;
            sF[wave][lane] = corr;
        }
        __syncthreads();

        // ---- rescale accumulator rows by the correction factor
        float cf[8];
#pragma unroll
        for (int r = 0; r < 8; ++r) cf[r] = sF[wave][r + 8 * half];
#pragma unroll
        for (int t = 0; t < 4; ++t)
#pragma unroll
            for (int r = 0; r < 8; ++r) o[t].f[r] *= cf[r];

        // ---- P as A-fragment (transpose via LDS), V fragments already resident
        AFrag pa;
#pragma unroll
        for (int j = 0; j < 8; ++j)
            pa.u[j] = sP[wave][ln * 16 + (j & 3) + ((j >> 2) << 3) + (half << 2)];
#pragma unroll
        for (int t = 0; t < 4; ++t)
            o[t].v = wmma_bf16(pa.v, vb[t].v, o[t].v);
        __syncthreads();
    }

    // ---- finalize: divide by running sum, store bf16
    if (lane < 16) sF[wave][lane] = (li > 0.f) ? (1.f / li) : 0.f;
    __syncthreads();
    float iv[8];
#pragma unroll
    for (int r = 0; r < 8; ++r) iv[r] = sF[wave][r + 8 * half];
#pragma unroll
    for (int t = 0; t < 4; ++t)
#pragma unroll
        for (int r = 0; r < 8; ++r) {
            int tok = b * SS + qbase + r + 8 * half;
            out[(size_t)tok * (NH_*HD_) + h * 64 + t * 16 + ln] = f32_to_bf16(o[t].f[r] * iv[r]);
        }
}

// ---------------- orchestration ----------------
extern "C" void kernel_launch(void* const* d_in, const int* in_sizes, int n_in,
                              void* d_out, int out_size, void* d_ws, size_t ws_size,
                              hipStream_t stream) {
    (void)in_sizes; (void)n_in; (void)out_size; (void)ws_size;
    const float* x         = (const float*)d_in[0];
    const int*   start_pos = (const int*)  d_in[1];
    const float* fcos      = (const float*)d_in[2];
    const float* fsin      = (const float*)d_in[3];
    const float* cache_k   = (const float*)d_in[4];
    const float* cache_v   = (const float*)d_in[5];
    const float* mask      = (const float*)d_in[6];
    const float* wq        = (const float*)d_in[7];
    const float* wk        = (const float*)d_in[8];
    const float* wv        = (const float*)d_in[9];
    const float* wo        = (const float*)d_in[10];
    const float* w1        = (const float*)d_in[11];
    const float* w2        = (const float*)d_in[12];
    const float* w3        = (const float*)d_in[13];
    const float* attn_nw   = (const float*)d_in[14];
    const float* ffn_nw    = (const float*)d_in[15];
    float* out = (float*)d_out;

    char* ws = (char*)d_ws;
    size_t off = 0;
    auto alloc = [&](size_t bytes) -> void* {
        void* p = ws + off;
        off += (bytes + 255) & ~(size_t)255;
        return p;
    };
    unsigned short* wq_bf = (unsigned short*)alloc((size_t)(NH_*HD_)*DD*2);
    unsigned short* wk_bf = (unsigned short*)alloc((size_t)(NKV_*HD_)*DD*2);
    unsigned short* wv_bf = (unsigned short*)alloc((size_t)(NKV_*HD_)*DD*2);
    unsigned short* wo_bf = (unsigned short*)alloc((size_t)DD*(NH_*HD_)*2);
    unsigned short* w1_bf = (unsigned short*)alloc((size_t)HID_*DD*2);
    unsigned short* w3_bf = (unsigned short*)alloc((size_t)HID_*DD*2);
    unsigned short* w2_bf = (unsigned short*)alloc((size_t)DD*HID_*2);
    unsigned short* h_bf  = (unsigned short*)alloc((size_t)TOK*DD*2);
    unsigned short* q_bf  = (unsigned short*)alloc((size_t)TOK*(NH_*HD_)*2);
    unsigned short* k_bf  = (unsigned short*)alloc((size_t)TOK*(NKV_*HD_)*2);
    unsigned short* v_bf  = (unsigned short*)alloc((size_t)TOK*(NKV_*HD_)*2);
    unsigned short* kc    = (unsigned short*)alloc((size_t)BB*MAXS_*NKV_*HD_*2);
    unsigned short* vt    = (unsigned short*)alloc((size_t)BB*MAXS_*NKV_*HD_*2);
    unsigned short* attn  = (unsigned short*)alloc((size_t)TOK*(NH_*HD_)*2);
    float*          h1    = (float*)         alloc((size_t)TOK*DD*4);
    unsigned short* g_bf  = (unsigned short*)alloc((size_t)TOK*DD*2);
    unsigned short* t1    = (unsigned short*)alloc((size_t)TOK*HID_*2);
    unsigned short* t3    = (unsigned short*)alloc((size_t)TOK*HID_*2);

    // 1) weights -> bf16
    cvt_bf16_kernel<<<4096, 256, 0, stream>>>(wq, wq_bf, (long long)(NH_*HD_)*DD);
    cvt_bf16_kernel<<<4096, 256, 0, stream>>>(wk, wk_bf, (long long)(NKV_*HD_)*DD);
    cvt_bf16_kernel<<<4096, 256, 0, stream>>>(wv, wv_bf, (long long)(NKV_*HD_)*DD);
    cvt_bf16_kernel<<<4096, 256, 0, stream>>>(wo, wo_bf, (long long)DD*(NH_*HD_));
    cvt_bf16_kernel<<<4096, 256, 0, stream>>>(w1, w1_bf, (long long)HID_*DD);
    cvt_bf16_kernel<<<4096, 256, 0, stream>>>(w3, w3_bf, (long long)HID_*DD);
    cvt_bf16_kernel<<<4096, 256, 0, stream>>>(w2, w2_bf, (long long)DD*HID_);

    // 2) attention rmsnorm -> bf16
    rmsnorm_kernel<<<TOK, 256, 0, stream>>>(x, attn_nw, h_bf, DD);

    // 3) Q/K/V projections (bf16 WMMA)
    gemm_bf16_kernel<0><<<dim3((NH_*HD_)/128, TOK/64), 256, 0, stream>>>(
        (const unsigned int*)h_bf, (const unsigned int*)wq_bf, q_bf, nullptr, nullptr, TOK, NH_*HD_, DD);
    gemm_bf16_kernel<0><<<dim3((NKV_*HD_)/128, TOK/64), 256, 0, stream>>>(
        (const unsigned int*)h_bf, (const unsigned int*)wk_bf, k_bf, nullptr, nullptr, TOK, NKV_*HD_, DD);
    gemm_bf16_kernel<0><<<dim3((NKV_*HD_)/128, TOK/64), 256, 0, stream>>>(
        (const unsigned int*)h_bf, (const unsigned int*)wv_bf, v_bf, nullptr, nullptr, TOK, NKV_*HD_, DD);

    // 4) RoPE (1/sqrt(64) folded into Q)
    {
        long long qp = (long long)TOK * NH_ * (HD_/2);
        rope_kernel<<<(unsigned)((qp + 255) / 256), 256, 0, stream>>>(q_bf, fcos, fsin, NH_, 0.125f, qp);
        long long kp = (long long)TOK * NKV_ * (HD_/2);
        rope_kernel<<<(unsigned)((kp + 255) / 256), 256, 0, stream>>>(k_bf, fcos, fsin, NKV_, 1.0f, kp);
    }

    // 5) KV caches (K compact, V transposed), bf16
    build_kcache_kernel <<<(BB*MAXS_*NKV_*HD_)/256, 256, 0, stream>>>(k_bf, cache_k, start_pos, kc);
    build_vcacheT_kernel<<<(BB*MAXS_*NKV_*HD_)/256, 256, 0, stream>>>(v_bf, cache_v, start_pos, vt);

    // 6) flash attention
    attn_kernel<<<BB * NH_ * (SS/64), 128, 0, stream>>>(
        (const unsigned int*)q_bf, (const unsigned int*)kc, (const unsigned int*)vt, mask, attn);

    // 7) output projection + residual -> h1 (f32)
    gemm_bf16_kernel<1><<<dim3(DD/128, TOK/64), 256, 0, stream>>>(
        (const unsigned int*)attn, (const unsigned int*)wo_bf, nullptr, h1, x, TOK, DD, NH_*HD_);

    // 8) ffn rmsnorm -> bf16
    rmsnorm_kernel<<<TOK, 256, 0, stream>>>(h1, ffn_nw, g_bf, DD);

    // 9) w1 / w3 GEMMs, silu-gate (in place into t1)
    gemm_bf16_kernel<0><<<dim3(HID_/128, TOK/64), 256, 0, stream>>>(
        (const unsigned int*)g_bf, (const unsigned int*)w1_bf, t1, nullptr, nullptr, TOK, HID_, DD);
    gemm_bf16_kernel<0><<<dim3(HID_/128, TOK/64), 256, 0, stream>>>(
        (const unsigned int*)g_bf, (const unsigned int*)w3_bf, t3, nullptr, nullptr, TOK, HID_, DD);
    silu_gate_kernel<<<4096, 256, 0, stream>>>(t1, t3, (long long)TOK * HID_);

    // 10) w2 GEMM + residual h1 -> final f32 output
    gemm_bf16_kernel<1><<<dim3(DD/128, TOK/64), 256, 0, stream>>>(
        (const unsigned int*)t1, (const unsigned int*)w2_bf, nullptr, out, h1, TOK, DD, HID_);
}
